// GNNmodel_11931419148813
// MI455X (gfx1250) — compile-verified, matrix operand order checked by reading, not run on hardware
//
#include <hip/hip_runtime.h>
#include <hip/hip_fp16.h>

typedef __attribute__((ext_vector_type(16))) _Float16 v16h;
typedef __attribute__((ext_vector_type(8)))  _Float16 v8h;
typedef __attribute__((ext_vector_type(8)))  float    v8f;

#define C1 100    // layer-1 width
#define C2 200    // layer-2 width
#define KP 128    // padded K for WMMA (100 -> 128)
#define NP 208    // padded layer-2 width (200 -> 13*16)
#define TILES_N (NP / 16)   // 13
#define KBLK (KP / 32)      // 4
#define MTILES 5            // M-tiles swept per wave (B fragments reused 5x)

__device__ __forceinline__ float silu_f(float v) { return v / (1.0f + __expf(-v)); }

// ---------------- utility ----------------
__global__ void zero_f32(float* p, long long n) {
  long long i = (long long)blockIdx.x * blockDim.x + threadIdx.x;
  long long stride = (long long)gridDim.x * blockDim.x;
  for (; i < n; i += stride) p[i] = 0.0f;
}

// ---------------- degree / norm ----------------
__global__ void deg_kernel(const int* __restrict__ dst, float* __restrict__ deg, int E) {
  int e = blockIdx.x * blockDim.x + threadIdx.x;
  if (e < E) atomicAdd(&deg[dst[e]], 1.0f);
}

__global__ void dinv_kernel(const float* __restrict__ deg, float* __restrict__ dinv, int N) {
  int n = blockIdx.x * blockDim.x + threadIdx.x;
  if (n < N) dinv[n] = rsqrtf(deg[n] + 1.0f);   // +1 self loop
}

// Layer-1 aggregation collapses to a scalar per node:
//   s1[d] += dinv[s]*dinv[d]*x[s]   (and cache norm[e] for layer 2)
__global__ void edge1_kernel(const int* __restrict__ src, const int* __restrict__ dst,
                             const float* __restrict__ x, const float* __restrict__ dinv,
                             float* __restrict__ normw, float* __restrict__ s1, int E) {
  int e = blockIdx.x * blockDim.x + threadIdx.x;
  if (e >= E) return;
  int s = src[e], d = dst[e];
  float nm = dinv[s] * dinv[d];
  normw[e] = nm;
  atomicAdd(&s1[d], nm * x[s]);
}

// h1[n,c] = silu(W1[c]*(s1[n] + x[n]*dinv^2) + b1[c]), stored f16 padded to KP
__global__ void h1_kernel(const float* __restrict__ x, const float* __restrict__ dinv,
                          const float* __restrict__ s1, const float* __restrict__ W1,
                          const float* __restrict__ b1, _Float16* __restrict__ h1h, int N) {
  long long idx = (long long)blockIdx.x * blockDim.x + threadIdx.x;
  int n = (int)(idx >> 7);
  int c = (int)(idx & 127);
  if (n >= N) return;
  float v = 0.0f;
  if (c < C1) {
    float dv = dinv[n];
    float sv = s1[n] + x[n] * dv * dv;
    v = silu_f(W1[c] * sv + b1[c]);
  }
  h1h[(long long)n * KP + c] = (_Float16)v;
}

// W2 [100x200] f32 -> B fragments in WMMA per-lane register layout:
//   bfrag[((tn*KBLK + kb)*32 + lane)*16 + i]
__global__ void w2frag_kernel(const float* __restrict__ W2, _Float16* __restrict__ bfrag) {
  int idx = blockIdx.x * blockDim.x + threadIdx.x;
  if (idx >= TILES_N * KBLK * 32 * 16) return;
  int i    = idx & 15;
  int lane = (idx >> 4) & 31;
  int kb   = (idx >> 9) & 3;
  int tn   = idx >> 11;
  int hs   = (lane < 16) ? 0 : 8;
  int n    = tn * 16 + (lane & 15);
  int k    = kb * 32 + ((i < 8) ? (hs + i) : (16 + hs + (i - 8)));
  float v = (k < C1 && n < C2) ? W2[k * C2 + n] : 0.0f;
  bfrag[idx] = (_Float16)v;
}

// ---------------- layer-2 GEMM via WMMA ----------------
__device__ __forceinline__ void wmma_one_tile(const _Float16* __restrict__ h1h,
                                              _Float16* __restrict__ h2h,
                                              const v16h bfr[KBLK],
                                              int tm, int tn, int lane, int hs, int l15) {
  const _Float16* arow = h1h + (long long)(tm * 16 + l15) * KP;
  v8f c = {};
#pragma unroll
  for (int kb = 0; kb < KBLK; kb++) {
    v8h a0 = *(const v8h*)(arow + kb * 32 + hs);       // K = kb*32+hs .. +7
    v8h a1 = *(const v8h*)(arow + kb * 32 + 16 + hs);  // K = kb*32+16+hs .. +7
    v16h a = __builtin_shufflevector(a0, a1, 0, 1, 2, 3, 4, 5, 6, 7,
                                             8, 9, 10, 11, 12, 13, 14, 15);
    c = __builtin_amdgcn_wmma_f32_16x16x32_f16(false, a, false, bfr[kb],
                                               (short)0, c, false, false);
  }
  int col = tn * 16 + l15;
  int rowbase = tm * 16 + ((lane < 16) ? 0 : 8);
#pragma unroll
  for (int r = 0; r < 8; r++)
    h2h[(long long)(rowbase + r) * NP + col] = (_Float16)c[r];
}

// One wave: all 4 B fragments held in registers, swept over MTILES row-tiles.
__global__ void wmma_gemm_kernel(const _Float16* __restrict__ h1h,
                                 const _Float16* __restrict__ bfrag,
                                 _Float16* __restrict__ h2h, int N) {
  int tilesM = N / 16;                           // N % 16 == 0 here
  int groups = (tilesM + MTILES - 1) / MTILES;
  int wave = threadIdx.x >> 5;
  int lane = threadIdx.x & 31;
  long long wid = (long long)blockIdx.x * (blockDim.x >> 5) + wave;
  long long total = (long long)groups * TILES_N;
  if (wid >= total) return;                      // wave-uniform
  int grp = (int)(wid / TILES_N);
  int tn  = (int)(wid % TILES_N);

  int hs  = (lane < 16) ? 0 : 8;
  int l15 = lane & 15;

  // B fragments: one contiguous 32B load per K-block (pre-swizzled)
  const _Float16* bp = bfrag + ((long long)(tn * KBLK) * 32 + lane) * 16;
  v16h bfr[KBLK];
#pragma unroll
  for (int kb = 0; kb < KBLK; kb++)
    bfr[kb] = *(const v16h*)(bp + (long long)kb * 32 * 16);

  int tm0 = grp * MTILES;
  if (tm0 + MTILES <= tilesM) {
    // fast path: no per-tile guard -> fully unrolled, B stays in registers
#pragma unroll
    for (int m = 0; m < MTILES; m++)
      wmma_one_tile(h1h, h2h, bfr, tm0 + m, tn, lane, hs, l15);
  } else {
    // generic tail (never taken when tilesM % MTILES == 0)
    int mEnd = tilesM - tm0;
    for (int m = 0; m < mEnd; m++)
      wmma_one_tile(h1h, h2h, bfr, tm0 + m, tn, lane, hs, l15);
  }
}

// ---------------- layer-2 aggregation (L2-resident gather + atomic scatter) ----------------
__global__ void edge2_kernel(const int* __restrict__ src, const int* __restrict__ dst,
                             const float* __restrict__ normw, const _Float16* __restrict__ h2h,
                             float* __restrict__ agg2, int E) {
  int e = blockIdx.x * (blockDim.x >> 6) + (threadIdx.x >> 6);  // 4 edges / 256-thread block
  int f = threadIdx.x & 63;
  if (e >= E) return;
  float nm = normw[e];
  const _Float16* hsrc = h2h + (long long)src[e] * NP;
  float* ad = agg2 + (long long)dst[e] * C2;
  for (int c = f; c < C2; c += 64)
    atomicAdd(&ad[c], nm * (float)hsrc[c]);
}

// h2 = silu(agg2 + h2raw*dinv^2 + b2); scatter into per-graph pool sums
__global__ void act2_pool_kernel(const _Float16* __restrict__ h2h, const float* __restrict__ agg2,
                                 const float* __restrict__ dinv, const float* __restrict__ b2,
                                 const int* __restrict__ batch, float* __restrict__ pool, int N) {
  long long idx = (long long)blockIdx.x * blockDim.x + threadIdx.x;
  long long total = (long long)N * C2;
  if (idx >= total) return;
  int n = (int)(idx / C2);
  int c = (int)(idx % C2);
  float dv = dinv[n];
  float v = agg2[idx] + (float)h2h[(long long)n * NP + c] * dv * dv + b2[c];
  atomicAdd(&pool[(long long)batch[n] * C2 + c], silu_f(v));
}

__global__ void cnt_kernel(const int* __restrict__ batch, float* __restrict__ cnt, int N) {
  int n = blockIdx.x * blockDim.x + threadIdx.x;
  if (n < N) atomicAdd(&cnt[batch[n]], 1.0f);
}

// ---------------- MLP head: one block per graph ----------------
__global__ void head_kernel(const float* __restrict__ pool, const float* __restrict__ cnt,
                            const float* __restrict__ Wl1, const float* __restrict__ bl1,
                            const float* __restrict__ Wl2, const float* __restrict__ bl2,
                            float* __restrict__ out) {
  __shared__ float sp[C2];
  __shared__ float hh[C1];
  int g = blockIdx.x;
  float inv = 1.0f / fmaxf(cnt[g], 1.0f);
  for (int c = threadIdx.x; c < C2; c += blockDim.x) sp[c] = pool[g * C2 + c] * inv;
  __syncthreads();
  for (int j = threadIdx.x; j < C1; j += blockDim.x) {
    float acc = bl1[j];
    for (int c = 0; c < C2; c++) acc += sp[c] * Wl1[c * C1 + j];
    hh[j] = silu_f(acc);
  }
  __syncthreads();
  if (threadIdx.x == 0) {
    float o = bl2[0];
    for (int j = 0; j < C1; j++) o += hh[j] * Wl2[j];
    out[g] = o;
  }
}

extern "C" void kernel_launch(void* const* d_in, const int* in_sizes, int n_in,
                              void* d_out, int out_size, void* d_ws, size_t ws_size,
                              hipStream_t stream) {
  const float* x   = (const float*)d_in[0];
  const int*   src = (const int*)d_in[1];
  const int*   dst = (const int*)d_in[2];
  const int*   bat = (const int*)d_in[3];
  const float* W1  = (const float*)d_in[4];
  const float* b1  = (const float*)d_in[5];
  const float* W2  = (const float*)d_in[6];
  const float* b2  = (const float*)d_in[7];
  const float* Wl1 = (const float*)d_in[8];
  const float* bl1 = (const float*)d_in[9];
  const float* Wl2 = (const float*)d_in[10];
  const float* bl2 = (const float*)d_in[11];
  float* out = (float*)d_out;

  int N = in_sizes[0];
  int E = in_sizes[1];
  int G = out_size;
  (void)n_in; (void)ws_size;

  // workspace carve-out (256B aligned slots)
  char* ws = (char*)d_ws;
  size_t off = 0;
  auto carve = [&](size_t bytes) -> char* {
    char* p = ws + off;
    off += (bytes + 255) & ~(size_t)255;
    return p;
  };
  float*    deg   = (float*)carve((size_t)N * 4);
  float*    dinv  = (float*)carve((size_t)N * 4);
  float*    s1    = (float*)carve((size_t)N * 4);
  float*    normw = (float*)carve((size_t)E * 4);
  float*    cnt   = (float*)carve((size_t)G * 4);
  float*    pool  = (float*)carve((size_t)G * C2 * 4);
  _Float16* bfrag = (_Float16*)carve((size_t)TILES_N * KBLK * 32 * 16 * 2);
  _Float16* h1h   = (_Float16*)carve((size_t)N * KP * 2);
  _Float16* h2h   = (_Float16*)carve((size_t)N * NP * 2);
  float*    agg2  = (float*)carve((size_t)N * C2 * 4);

  const int T = 256;
  // zero accumulators (ws is poisoned)
  zero_f32<<<(N + T - 1) / T, T, 0, stream>>>(deg, N);
  zero_f32<<<(N + T - 1) / T, T, 0, stream>>>(s1, N);
  zero_f32<<<(int)(((long long)N * C2 + T - 1) / T), T, 0, stream>>>(agg2, (long long)N * C2);
  zero_f32<<<(G * C2 + T - 1) / T, T, 0, stream>>>(pool, (long long)G * C2);
  zero_f32<<<1, 64, 0, stream>>>(cnt, G);

  // degrees + normalization
  deg_kernel<<<(E + T - 1) / T, T, 0, stream>>>(dst, deg, E);
  dinv_kernel<<<(N + T - 1) / T, T, 0, stream>>>(deg, dinv, N);

  // layer 1 (rank-1 trick): scalar edge aggregation, then activation into f16
  edge1_kernel<<<(E + T - 1) / T, T, 0, stream>>>(src, dst, x, dinv, normw, s1, E);
  h1_kernel<<<(int)(((long long)N * KP + T - 1) / T), T, 0, stream>>>(x, dinv, s1, W1, b1, h1h, N);

  // layer 2 GEMM on tensor cores (B pre-swizzled into WMMA lane layout)
  w2frag_kernel<<<(TILES_N * KBLK * 32 * 16 + T - 1) / T, T, 0, stream>>>(W2, bfrag);
  int tilesM = N / 16;
  long long totalWaves = (long long)((tilesM + MTILES - 1) / MTILES) * TILES_N;
  int wavesPerBlock = T / 32;
  wmma_gemm_kernel<<<(int)((totalWaves + wavesPerBlock - 1) / wavesPerBlock), T, 0, stream>>>(
      h1h, bfrag, h2h, N);

  // layer 2 aggregation + activation + pooling
  edge2_kernel<<<(E + 3) / 4, T, 0, stream>>>(src, dst, normw, h2h, agg2, E);
  act2_pool_kernel<<<(int)(((long long)N * C2 + T - 1) / T), T, 0, stream>>>(
      h2h, agg2, dinv, b2, bat, pool, N);
  cnt_kernel<<<(N + T - 1) / T, T, 0, stream>>>(bat, cnt, N);

  // MLP head
  head_kernel<<<G, 128, 0, stream>>>(pool, cnt, Wl1, bl1, Wl2, bl2, out);
}
